// CDFE_81415400063357
// MI455X (gfx1250) — matrix operand order repaired and checked by reading, not compiled, not run on hardware
//
#include <hip/hip_runtime.h>
#include <hip/hip_bf16.h>

typedef __attribute__((ext_vector_type(16))) _Float16 v16h;
typedef __attribute__((ext_vector_type(8)))  _Float16 v8h;
typedef __attribute__((ext_vector_type(8)))  float    v8f;

#define TT 16
#define HH 48
#define WW 48
#define VS (TT*HH*WW)      // 36864 key tokens
#define SS (HH*WW)         // 2304 query tokens
#define CC 64
#define DD 6
#define LN_EPS 1e-5f

#define WAVES 16                       // 512 threads/block -> 2304 waves total
#define KT_TOTAL (VS/16)               // 2304 key tiles of 16
#define KT_PER_WAVE (KT_TOTAL/WAVES)   // 144

// ---------------------------------------------------------------------------
// Projection (64 -> 6) + LayerNorm over dim 6, emit f16 tokens padded to 8
// halves (elements 6,7 = 0) so the attention kernel can do 128-bit loads.
// Channel-major input: x[c*ntok + v].
// ---------------------------------------------------------------------------
__global__ __launch_bounds__(256) void proj_ln_kernel(
    const float* __restrict__ x,      // (CC, ntok)
    const float* __restrict__ wt,     // (DD, CC) row-major
    const float* __restrict__ bias,   // (DD,)
    const float* __restrict__ gamma,  // (DD,)
    const float* __restrict__ beta,   // (DD,)
    _Float16* __restrict__ out,       // (ntok, 8)
    int ntok)
{
    __shared__ float wl[CC*DD];
    __shared__ float sb[DD], sg[DD], se[DD];
    int tid = threadIdx.x;
    if (tid < CC*DD) {
        int c = tid / DD, d = tid % DD;
        wl[tid] = wt[d*CC + c];            // store as [c][d]
    }
    if (tid < DD) { sb[tid] = bias[tid]; sg[tid] = gamma[tid]; se[tid] = beta[tid]; }
    __syncthreads();

    int v = blockIdx.x * 256 + tid;
    if (v >= ntok) return;

    float acc[DD];
    #pragma unroll
    for (int d = 0; d < DD; ++d) acc[d] = 0.f;
    for (int c = 0; c < CC; ++c) {
        float xv = x[c*ntok + v];          // coalesced across threads
        #pragma unroll
        for (int d = 0; d < DD; ++d) acc[d] = fmaf(xv, wl[c*DD + d], acc[d]);
    }
    float mu = 0.f;
    #pragma unroll
    for (int d = 0; d < DD; ++d) { acc[d] += sb[d]; mu += acc[d]; }
    mu *= (1.f/DD);
    float var = 0.f;
    #pragma unroll
    for (int d = 0; d < DD; ++d) { float t = acc[d] - mu; var = fmaf(t, t, var); }
    var *= (1.f/DD);
    float rs = rsqrtf(var + LN_EPS);
    #pragma unroll
    for (int d = 0; d < DD; ++d)
        out[v*8 + d] = (_Float16)((acc[d] - mu) * rs * sg[d] + se[d]);
    out[v*8 + 6] = (_Float16)0.f;
    out[v*8 + 7] = (_Float16)0.f;
}

// ---------------------------------------------------------------------------
// Fused attention: S = Q.K^T via v_wmma_f32_16x16x32_f16 (K padded 6->32),
// streaming softmax (no max subtraction needed: |logit| <= ~6 after LN),
// weighted sum of analytic g_vol, final flow output. One block = one 16-query
// tile; 16 waves split the 2304 key tiles and combine partials in LDS.
// ---------------------------------------------------------------------------
__global__ __launch_bounds__(512) void attn_flow_kernel(
    const _Float16* __restrict__ Q,   // (SS, 8)
    const _Float16* __restrict__ K,   // (VS, 8)
    float* __restrict__ outp)         // (3, SS)
{
    __shared__ float red[WAVES][16][4];

    int tid  = threadIdx.x;
    int wid  = tid >> 5;
    int lane = tid & 31;
    int n    = lane & 15;     // column (key within tile) for B/C/D
    int half = lane >> 4;

    // A-matrix (16x32 f16): lane holds row m=n; halves 0..7 <- Q token
    // (6 valid + 2 stored zeros), halves 8..15 = 0 so every K>=6 product
    // vanishes once B zeroes its lanes 16..31 (A holds Q in both lane halves).
    int qtok = blockIdx.x * 16 + n;
    v8h q8 = *(const v8h*)(Q + (size_t)qtok * 8);
    v16h a;
    #pragma unroll
    for (int e = 0; e < 8; ++e) a[e] = q8[e];
    #pragma unroll
    for (int e = 8; e < 16; ++e) a[e] = (_Float16)0.f;

    float sacc[8], gx[8], gy[8], gz[8];
    #pragma unroll
    for (int j = 0; j < 8; ++j) { sacc[j] = 0.f; gx[j] = 0.f; gy[j] = 0.f; gz[j] = 0.f; }

    // Incremental (t,h,w) for this lane's key token; advances by 16 per tile.
    int kt0  = wid * KT_PER_WAVE;
    int vtok = kt0 * 16 + n;
    int ti = vtok / (HH*WW);
    int rem = vtok % (HH*WW);
    int hi = rem / WW;
    int wi = rem % WW;

    for (int kt = 0; kt < KT_PER_WAVE; ++kt) {
        int ktok = (kt0 + kt) * 16 + n;
        v8h k8 = *(const v8h*)(K + (size_t)ktok * 8);
        v16h b;
        #pragma unroll
        for (int e = 0; e < 8; ++e) b[e] = half ? (_Float16)0.f : k8[e];
        #pragma unroll
        for (int e = 8; e < 16; ++e) b[e] = (_Float16)0.f;

        v8f c = {};
        c = __builtin_amdgcn_wmma_f32_16x16x32_f16(
                /*neg_a=*/false, a, /*neg_b=*/false, b,
                /*c_mod=*/(short)0, c, /*reuse_a=*/false, /*reuse_b=*/false);

        float gt = (float)ti - 0.5f*(TT-1);
        float gh = (float)hi - 0.5f*(HH-1);
        float gw = (float)wi - 0.5f*(WW-1);
        #pragma unroll
        for (int j = 0; j < 8; ++j) {
            float e = __expf(c[j]);        // D[j] = S[row j+8*half][col n]
            sacc[j] += e;
            gx[j] = fmaf(e, gt, gx[j]);
            gy[j] = fmaf(e, gh, gy[j]);
            gz[j] = fmaf(e, gw, gz[j]);
        }
        wi += 16;
        if (wi >= WW) { wi -= WW; hi += 1; if (hi >= HH) { hi -= HH; ti += 1; } }
    }

    // Reduce over key columns n (16 lanes within each half-wave).
    #pragma unroll
    for (int m = 1; m < 16; m <<= 1) {
        #pragma unroll
        for (int j = 0; j < 8; ++j) {
            sacc[j] += __shfl_xor(sacc[j], m, 32);
            gx[j]   += __shfl_xor(gx[j],   m, 32);
            gy[j]   += __shfl_xor(gy[j],   m, 32);
            gz[j]   += __shfl_xor(gz[j],   m, 32);
        }
    }
    if (n == 0) {
        #pragma unroll
        for (int j = 0; j < 8; ++j) {
            int mrow = j + 8*half;
            red[wid][mrow][0] = sacc[j];
            red[wid][mrow][1] = gx[j];
            red[wid][mrow][2] = gy[j];
            red[wid][mrow][3] = gz[j];
        }
    }
    __syncthreads();

    if (tid < 16) {
        float s = 0.f, x = 0.f, y = 0.f, z = 0.f;
        #pragma unroll
        for (int wv = 0; wv < WAVES; ++wv) {
            s += red[wv][tid][0];
            x += red[wv][tid][1];
            y += red[wv][tid][2];
            z += red[wv][tid][3];
        }
        float inv = 1.f / s;
        int st = blockIdx.x * 16 + tid;
        int h = st / WW, w = st % WW;
        outp[0*SS + st] = x * inv;                               // g_slice t-comp = 0
        outp[1*SS + st] = y * inv - ((float)h - 0.5f*(HH-1));
        outp[2*SS + st] = z * inv - ((float)w - 0.5f*(WW-1));
    }
}

// ---------------------------------------------------------------------------
extern "C" void kernel_launch(void* const* d_in, const int* in_sizes, int n_in,
                              void* d_out, int out_size, void* d_ws, size_t ws_size,
                              hipStream_t stream) {
    (void)in_sizes; (void)n_in; (void)out_size; (void)ws_size;
    const float* vol  = (const float*)d_in[0];
    const float* slc  = (const float*)d_in[1];
    const float* w2d  = (const float*)d_in[2];
    const float* b2d  = (const float*)d_in[3];
    const float* g2d  = (const float*)d_in[4];
    const float* be2d = (const float*)d_in[5];
    const float* w3d  = (const float*)d_in[6];
    const float* b3d  = (const float*)d_in[7];
    const float* g3d  = (const float*)d_in[8];
    const float* be3d = (const float*)d_in[9];
    float* out = (float*)d_out;

    _Float16* K16 = (_Float16*)d_ws;
    _Float16* Q16 = (_Float16*)((char*)d_ws + (size_t)VS * 8 * sizeof(_Float16));

    proj_ln_kernel<<<VS/256, 256, 0, stream>>>(vol, w3d, b3d, g3d, be3d, K16, VS);
    proj_ln_kernel<<<SS/256, 256, 0, stream>>>(slc, w2d, b2d, g2d, be2d, Q16, SS);
    attn_flow_kernel<<<SS/16, 512, 0, stream>>>(Q16, K16, out);
}
